// GNN_LSTM_36060545417515
// MI455X (gfx1250) — matrix operand order repaired and checked by reading, not compiled, not run on hardware
//
#include <hip/hip_runtime.h>

#define B_    64
#define T_    128
#define N_    16
#define F_    90
#define FP_   96          // F padded to multiple of 32 for bf16 WMMA K
#define GOUT_ 256
#define H1_   128
#define H2_   256
#define E_    4096
#define NN_   1024        // B_*N_
#define M_GCN (NN_*T_)    // 131072 rows of the GCN GEMM
#define SEQ_I (N_*GOUT_)  // 4096 LSTM1 input width
#define MB_   (T_*B_)     // 8192 rows of LSTM1 input GEMM
#define G1_   (4*H1_)     // 512 gate width LSTM1
#define G2_   (4*H2_)     // 1024 gate width LSTM2

typedef __bf16 bf16;
typedef __attribute__((ext_vector_type(16))) bf16  v16bf;
typedef __attribute__((ext_vector_type(8)))  bf16  v8bf;
typedef __attribute__((ext_vector_type(8)))  float v8f;

__device__ __forceinline__ v8f splat8(float x) {
  v8f v;
#pragma unroll
  for (int i = 0; i < 8; ++i) v[i] = x;
  return v;
}

// A fragment (16x32 bf16, M x K), A row-major with leading dim ld.
// Layout used here: lane holds row M=lane&15; k = (lane>>4)*16 + {0..15} (contiguous, matches B).
__device__ __forceinline__ v16bf load_a_rm(const bf16* A, int rowBase, int ld, int k0) {
  const int lane = threadIdx.x & 31;
  const bf16* p = A + (size_t)(rowBase + (lane & 15)) * ld + k0 + ((lane >> 4) << 4);
  v8bf lo = *(const v8bf*)(p);
  v8bf hi = *(const v8bf*)(p + 8);
  return __builtin_shufflevector(lo, hi, 0,1,2,3,4,5,6,7,8,9,10,11,12,13,14,15);
}

// B fragment (32x16 bf16, K x N) taken as W^T where W is row-major [Ntot, K].
// Layout used here: lane holds col N=lane&15; k = (lane>>4)*16 + {0..15} (contiguous -> 32B load).
__device__ __forceinline__ v16bf load_b_wt(const bf16* W, int nBase, int ld, int k0) {
  const int lane = threadIdx.x & 31;
  return *(const v16bf*)(W + (size_t)(nBase + (lane & 15)) * ld + k0 + ((lane >> 4) << 4));
}

__device__ __forceinline__ v8f wmma_bf(v16bf a, v16bf b, v8f c) {
  // (neg_a, A, neg_b, B, c_mod, C, reuse_a, reuse_b)
  return __builtin_amdgcn_wmma_f32_16x16x32_bf16(false, a, false, b, (short)0, c, false, false);
}

__device__ __forceinline__ float sigm(float x) { return 1.0f / (1.0f + __expf(-x)); }

// ---------------- GCN graph prep ----------------

__global__ void k_init_deg(float* deg) {
  int i = blockIdx.x * blockDim.x + threadIdx.x;
  if (i < NN_) deg[i] = 1.0f;  // self-loop
}

__global__ void k_zero_f32(float* p, int n) {
  for (int i = blockIdx.x * blockDim.x + threadIdx.x; i < n; i += gridDim.x * blockDim.x)
    p[i] = 0.0f;
}

__global__ void k_deg(const long long* ei, float* deg) {
  int e = blockIdx.x * blockDim.x + threadIdx.x;
  if (e < E_) atomicAdd(&deg[(int)ei[E_ + e]], 1.0f);
}

__global__ void k_dinv(float* deg) {
  int i = blockIdx.x * blockDim.x + threadIdx.x;
  if (i < NN_) deg[i] = rsqrtf(deg[i]);  // deg >= 1 always (self-loop)
}

// One block per (edge or self-loop); scatter norm * x[src] into agg[dst] over T*F.
__global__ void k_scatter(const float* x, const long long* ei, const float* dinv, float* agg) {
  int e = blockIdx.x;
  int src, dst;
  if (e < E_) { src = (int)ei[e]; dst = (int)ei[E_ + e]; }
  else        { src = dst = e - E_; }
  float norm = dinv[src] * dinv[dst];
  int bs = src >> 4, is = src & 15;
  const float* xs = x + ((size_t)bs * T_ * N_ + is) * F_;  // x[bs][t][is][f]
  float* ad = agg + (size_t)dst * T_ * F_;
  for (int j = threadIdx.x; j < T_ * F_; j += blockDim.x) {
    int t = j / F_, f = j - t * F_;
    atomicAdd(&ad[j], norm * xs[(size_t)t * N_ * F_ + f]);
  }
}

// ---------------- precision conversion ----------------

__global__ void k_cvt_bf16(const float* s, bf16* d, int n) {
  for (int i = blockIdx.x * blockDim.x + threadIdx.x; i < n; i += gridDim.x * blockDim.x)
    d[i] = (bf16)s[i];
}

// gcn_w [F_,GOUT_] -> transposed+padded bf16 [GOUT_, FP_] so B loads are contiguous
__global__ void k_prep_gcnw(const float* w, bf16* wt) {
  int i = blockIdx.x * blockDim.x + threadIdx.x;
  if (i < GOUT_ * FP_) {
    int o = i / FP_, k = i - o * FP_;
    wt[i] = (k < F_) ? (bf16)w[k * GOUT_ + o] : (bf16)0.0f;
  }
}

// agg f32 [M_GCN, F_] -> bf16 [M_GCN, FP_] zero-padded
__global__ void k_pad_agg(const float* agg, bf16* aggb) {
  for (int i = blockIdx.x * blockDim.x + threadIdx.x; i < M_GCN * FP_;
       i += gridDim.x * blockDim.x) {
    int m = i / FP_, k = i - m * FP_;
    aggb[i] = (k < F_) ? (bf16)agg[(size_t)m * F_ + k] : (bf16)0.0f;
  }
}

// ---------------- GEMM 1: GCN weight matmul -> seq (bf16, LSTM layout) ----------------
// seq[t*B + b][i*GOUT + o] = agg[(b*16+i)*T + t] @ gcn_w + gcn_b
__global__ __launch_bounds__(256) void k_gemm_gcn(const bf16* __restrict__ aggb,
                                                  const bf16* __restrict__ wt,
                                                  const float* __restrict__ bias,
                                                  bf16* __restrict__ seqb) {
  int wave = threadIdx.x >> 5;
  int id = blockIdx.x * 8 + wave;      // 32768 wave jobs: mt[0,8192) x ng[0,4)
  int mt = id >> 2, ng = id & 3;
  int lane = threadIdx.x & 31;
  int col0 = lane & 15, hi8 = (lane >> 4) * 8;
  v8f acc[4];
#pragma unroll
  for (int j = 0; j < 4; ++j) acc[j] = splat8(bias[(ng * 4 + j) * 16 + col0]);
#pragma unroll
  for (int ks = 0; ks < FP_ / 32; ++ks) {
    v16bf a = load_a_rm(aggb, mt * 16, FP_, ks * 32);
#pragma unroll
    for (int j = 0; j < 4; ++j) {
      v16bf b = load_b_wt(wt, (ng * 4 + j) * 16, FP_, ks * 32);
      acc[j] = wmma_bf(a, b, acc[j]);
    }
  }
#pragma unroll
  for (int j = 0; j < 4; ++j) {
    int ncol = (ng * 4 + j) * 16 + col0;
#pragma unroll
    for (int r = 0; r < 8; ++r) {
      int m = mt * 16 + hi8 + r;          // m = node*T + t
      int node = m >> 7, t = m & 127;
      int bb = node >> 4, ii = node & 15;
      seqb[((size_t)(t * B_ + bb)) * SEQ_I + ii * GOUT_ + ncol] = (bf16)acc[j][r];
    }
  }
}

// ---------------- GEMM 2: LSTM1 input matmul -> xg1 (f32) ----------------
// xg1[t*B+b][g] = seq @ wih1^T + bih1 + bhh1   (M=8192, N=512, K=4096)
__global__ __launch_bounds__(256) void k_gemm_xg1(const bf16* __restrict__ seqb,
                                                  const bf16* __restrict__ wih1,
                                                  const float* __restrict__ bih,
                                                  const float* __restrict__ bhh,
                                                  float* __restrict__ xg1) {
  int wave = threadIdx.x >> 5;
  int id = blockIdx.x * 8 + wave;      // 4096 wave jobs: mt[0,512) x ng[0,8)
  int mt = id >> 3, ng = id & 7;
  int lane = threadIdx.x & 31;
  int col0 = lane & 15, hi8 = (lane >> 4) * 8;
  v8f acc[4];
#pragma unroll
  for (int j = 0; j < 4; ++j) {
    int n = (ng * 4 + j) * 16 + col0;
    acc[j] = splat8(bih[n] + bhh[n]);
  }
  for (int ks = 0; ks < SEQ_I / 32; ++ks) {   // 128 k-steps
    v16bf a = load_a_rm(seqb, mt * 16, SEQ_I, ks * 32);
#pragma unroll
    for (int j = 0; j < 4; ++j) {
      v16bf b = load_b_wt(wih1, (ng * 4 + j) * 16, SEQ_I, ks * 32);
      acc[j] = wmma_bf(a, b, acc[j]);
    }
  }
#pragma unroll
  for (int j = 0; j < 4; ++j) {
    int n = (ng * 4 + j) * 16 + col0;
#pragma unroll
    for (int r = 0; r < 8; ++r)
      xg1[(size_t)(mt * 16 + hi8 + r) * G1_ + n] = acc[j][r];
  }
}

// ---------------- fused recurrent LSTM1+LSTM2 (single WGP, 32 waves) ----------------
// Each wave owns gate tiles; c-state lives in VGPRs; h-state in LDS (h1 double-buffered).
__global__ __launch_bounds__(1024) void k_lstm(const float* __restrict__ xg1,
                                               const bf16* __restrict__ whh1,
                                               const bf16* __restrict__ wih2,
                                               const bf16* __restrict__ whh2,
                                               const float* __restrict__ bih2,
                                               const float* __restrict__ bhh2,
                                               float* __restrict__ hT) {
  __shared__ __attribute__((aligned(32))) bf16 h1bf[2][B_][H1_];  // 32 KB
  __shared__ __attribute__((aligned(32))) bf16 h2bf[B_][H2_];     // 32 KB
  const int tid = threadIdx.x;
  const int wave = tid >> 5;
  const int lane = tid & 31;
  const int col0 = lane & 15, hi8 = (lane >> 4) * 8;
  for (int i = tid; i < 2 * B_ * H1_; i += 1024) ((bf16*)h1bf)[i] = (bf16)0.0f;
  for (int i = tid; i < B_ * H2_; i += 1024) ((bf16*)h2bf)[i] = (bf16)0.0f;
  v8f c1 = splat8(0.0f);
  v8f c2[2] = { splat8(0.0f), splat8(0.0f) };
  const int mt1 = wave >> 3, nt1 = wave & 7;  // LSTM1: 4 m-tiles x 8 n-tiles per quadrant
  __syncthreads();
  int p = 0;
  for (int t = 0; t < T_; ++t) {
    // ---- LSTM1: gates = xg1[t] + h1_{t-1} @ whh1^T ----
    v8f acc[4];
#pragma unroll
    for (int g = 0; g < 4; ++g) {
      v8f a;
#pragma unroll
      for (int r = 0; r < 8; ++r)
        a[r] = xg1[(size_t)(t * B_ + mt1 * 16 + hi8 + r) * G1_ + g * H1_ + nt1 * 16 + col0];
      acc[g] = a;
    }
#pragma unroll
    for (int ks = 0; ks < H1_ / 32; ++ks) {
      v16bf a = load_a_rm(&h1bf[p][0][0], mt1 * 16, H1_, ks * 32);
#pragma unroll
      for (int g = 0; g < 4; ++g) {
        v16bf b = load_b_wt(whh1, g * H1_ + nt1 * 16, H1_, ks * 32);
        acc[g] = wmma_bf(a, b, acc[g]);
      }
    }
#pragma unroll
    for (int r = 0; r < 8; ++r) {
      float c = sigm(acc[1][r]) * c1[r] + sigm(acc[0][r]) * tanhf(acc[2][r]);
      c1[r] = c;
      float h = sigm(acc[3][r]) * tanhf(c);
      h1bf[p ^ 1][mt1 * 16 + hi8 + r][nt1 * 16 + col0] = (bf16)h;
    }
    __syncthreads();
    // ---- LSTM2: gates = h1_t @ wih2^T + h2_{t-1} @ whh2^T + biases (2 tiles/wave) ----
    v8f acc2[2][4];
#pragma unroll
    for (int pp = 0; pp < 2; ++pp) {
      int q = wave * 2 + pp;
      int mt = q >> 4, nt = q & 15;
#pragma unroll
      for (int g = 0; g < 4; ++g) {
        int n = g * H2_ + nt * 16 + col0;
        acc2[pp][g] = splat8(bih2[n] + bhh2[n]);
      }
#pragma unroll
      for (int ks = 0; ks < H1_ / 32; ++ks) {
        v16bf a = load_a_rm(&h1bf[p ^ 1][0][0], mt * 16, H1_, ks * 32);
#pragma unroll
        for (int g = 0; g < 4; ++g) {
          v16bf b = load_b_wt(wih2, g * H2_ + nt * 16, H1_, ks * 32);
          acc2[pp][g] = wmma_bf(a, b, acc2[pp][g]);
        }
      }
#pragma unroll
      for (int ks = 0; ks < H2_ / 32; ++ks) {
        v16bf a = load_a_rm(&h2bf[0][0], mt * 16, H2_, ks * 32);
#pragma unroll
        for (int g = 0; g < 4; ++g) {
          v16bf b = load_b_wt(whh2, g * H2_ + nt * 16, H2_, ks * 32);
          acc2[pp][g] = wmma_bf(a, b, acc2[pp][g]);
        }
      }
    }
    __syncthreads();   // all reads of h2bf complete before overwrite
#pragma unroll
    for (int pp = 0; pp < 2; ++pp) {
      int q = wave * 2 + pp;
      int mt = q >> 4, nt = q & 15;
#pragma unroll
      for (int r = 0; r < 8; ++r) {
        float c = sigm(acc2[pp][1][r]) * c2[pp][r] + sigm(acc2[pp][0][r]) * tanhf(acc2[pp][2][r]);
        c2[pp][r] = c;
        float h = sigm(acc2[pp][3][r]) * tanhf(c);
        h2bf[mt * 16 + hi8 + r][nt * 16 + col0] = (bf16)h;
        if (t == T_ - 1) hT[(mt * 16 + hi8 + r) * H2_ + nt * 16 + col0] = h;
      }
    }
    __syncthreads();
    p ^= 1;
  }
}

// ---------------- regressor head (tiny, fp32 VALU) ----------------
__global__ __launch_bounds__(256) void k_head(const float* __restrict__ hT,
                                              const float* w1, const float* b1,
                                              const float* w2, const float* b2,
                                              const float* w3, const float* b3,
                                              float* __restrict__ out) {
  __shared__ float l1[B_][64];
  __shared__ float l2[B_][32];
  int tid = threadIdx.x;
  for (int i = tid; i < B_ * 64; i += 256) {
    int bb = i >> 6, o = i & 63;
    float s = b1[o];
    for (int k = 0; k < H2_; ++k) s += hT[bb * H2_ + k] * w1[o * H2_ + k];
    l1[bb][o] = (s > 0.0f) ? s : 0.01f * s;
  }
  __syncthreads();
  for (int i = tid; i < B_ * 32; i += 256) {
    int bb = i >> 5, o = i & 31;
    float s = b2[o];
    for (int k = 0; k < 64; ++k) s += l1[bb][k] * w2[o * 64 + k];
    l2[bb][o] = (s > 0.0f) ? s : 0.01f * s;
  }
  __syncthreads();
  for (int i = tid; i < B_ * 5; i += 256) {
    int bb = i / 5, o = i - bb * 5;
    float s = b3[o];
    for (int k = 0; k < 32; ++k) s += l2[bb][k] * w3[o * 32 + k];
    out[i] = 4.0f / (1.0f + __expf(-s)) + 1.0f;
  }
}

extern "C" void kernel_launch(void* const* d_in, const int* in_sizes, int n_in,
                              void* d_out, int out_size, void* d_ws, size_t ws_size,
                              hipStream_t stream) {
  (void)in_sizes; (void)n_in; (void)out_size; (void)ws_size;
  const float*     x      = (const float*)d_in[0];
  const long long* ei     = (const long long*)d_in[1];   // int64 edge_index [2,E]
  const float*     gcn_w  = (const float*)d_in[2];
  const float*     gcn_b  = (const float*)d_in[3];
  const float*     l1_wih = (const float*)d_in[4];
  const float*     l1_whh = (const float*)d_in[5];
  const float*     l1_bih = (const float*)d_in[6];
  const float*     l1_bhh = (const float*)d_in[7];
  const float*     l2_wih = (const float*)d_in[8];
  const float*     l2_whh = (const float*)d_in[9];
  const float*     l2_bih = (const float*)d_in[10];
  const float*     l2_bhh = (const float*)d_in[11];
  const float*     rw1 = (const float*)d_in[12];
  const float*     rb1 = (const float*)d_in[13];
  const float*     rw2 = (const float*)d_in[14];
  const float*     rb2 = (const float*)d_in[15];
  const float*     rw3 = (const float*)d_in[16];
  const float*     rb3 = (const float*)d_in[17];
  float* out = (float*)d_out;

  char* base = (char*)d_ws;
  size_t off = 0;
  auto carve = [&](size_t bytes) -> char* {
    char* p = base + off;
    off = (off + bytes + 255) & ~(size_t)255;
    return p;
  };
  float* deg  = (float*)carve((size_t)NN_ * 4);                 // degree -> dinv in place
  float* agg  = (float*)carve((size_t)M_GCN * F_ * 4);          // A@X, fp32
  bf16*  aggb = (bf16*) carve((size_t)M_GCN * FP_ * 2);         // padded bf16
  bf16*  seqb = (bf16*) carve((size_t)MB_ * SEQ_I * 2);         // GCN out in LSTM layout
  float* xg1  = (float*)carve((size_t)MB_ * G1_ * 4);           // LSTM1 input gates
  bf16*  wgT  = (bf16*) carve((size_t)GOUT_ * FP_ * 2);
  bf16*  w1ih = (bf16*) carve((size_t)G1_ * SEQ_I * 2);
  bf16*  w1hh = (bf16*) carve((size_t)G1_ * H1_ * 2);
  bf16*  w2ih = (bf16*) carve((size_t)G2_ * H1_ * 2);
  bf16*  w2hh = (bf16*) carve((size_t)G2_ * H2_ * 2);
  float* hT   = (float*)carve((size_t)B_ * H2_ * 4);

  k_init_deg<<<4, 256, 0, stream>>>(deg);
  k_zero_f32<<<4096, 256, 0, stream>>>(agg, M_GCN * F_);
  k_deg<<<(E_ + 255) / 256, 256, 0, stream>>>(ei, deg);
  k_dinv<<<4, 256, 0, stream>>>(deg);
  k_scatter<<<E_ + NN_, 256, 0, stream>>>(x, ei, deg, agg);
  k_cvt_bf16<<<2048, 256, 0, stream>>>(l1_wih, w1ih, G1_ * SEQ_I);
  k_cvt_bf16<<<256, 256, 0, stream>>>(l1_whh, w1hh, G1_ * H1_);
  k_cvt_bf16<<<512, 256, 0, stream>>>(l2_wih, w2ih, G2_ * H1_);
  k_cvt_bf16<<<1024, 256, 0, stream>>>(l2_whh, w2hh, G2_ * H2_);
  k_prep_gcnw<<<(GOUT_ * FP_ + 255) / 256, 256, 0, stream>>>(gcn_w, wgT);
  k_pad_agg<<<4096, 256, 0, stream>>>(agg, aggb);
  k_gemm_gcn<<<4096, 256, 0, stream>>>(aggb, wgT, gcn_b, seqb);
  k_gemm_xg1<<<512, 256, 0, stream>>>(seqb, w1ih, l1_bih, l1_bhh, xg1);
  k_lstm<<<1, 1024, 0, stream>>>(xg1, w1hh, w2ih, w2hh, l2_bih, l2_bhh, hT);
  k_head<<<1, 256, 0, stream>>>(hT, rw1, rb1, rw2, rb2, rw3, rb3, out);
}
